// LSTM_36352603193585
// MI455X (gfx1250) — matrix operand order repaired and checked by reading, not compiled, not run on hardware
//
#include <hip/hip_runtime.h>
#include <hip/hip_bf16.h>
#include <math.h>

// ---------------- problem constants ----------------
#define VOCAB 30000
#define NTAGS 20
#define EMB   200
#define HID   128
#define BB    128
#define SS    512
#define G4    (4*HID)        // 512 gate width
#define KPAD  224            // EMB padded to 7 k-tiles of 32
#define NROWS (SS*BB)        // 65536 (t-major rows: r = t*B + b)

typedef __attribute__((ext_vector_type(16))) _Float16 v16h;
typedef __attribute__((ext_vector_type(8)))  _Float16 v8h;
typedef __attribute__((ext_vector_type(8)))  float    v8f;

__device__ __forceinline__ v8f wmma16(v16h a, v16h b, v8f c) {
  // v_wmma_f32_16x16x32_f16 : (neg_a, A, neg_b, B, c_mod, C, reuse_a, reuse_b)
  return __builtin_amdgcn_wmma_f32_16x16x32_f16(false, a, false, b, (short)0, c, false, false);
}

__device__ __forceinline__ v8f vzero8() {
  v8f z;
#pragma unroll
  for (int i = 0; i < 8; ++i) z[i] = 0.0f;
  return z;
}

// A-fragment (16-bit 16x32): lane row = M, two contiguous 8-half chunks
// lanes 0-15: K = k0+0..7 and k0+16..23 ; lanes 16-31: K = k0+8..15 and k0+24..31
__device__ __forceinline__ v16h load_afrag(const _Float16* rowp, int lane) {
  const _Float16* p = rowp + ((lane >> 4) << 3);  // +8 halves for upper half-wave
  v8h lo = *(const v8h*)(p);
  v8h hi = *(const v8h*)(p + 16);
  v16h a;
#pragma unroll
  for (int i = 0; i < 8; ++i) { a[i] = lo[i]; a[i + 8] = hi[i]; }
  return a;
}

// B-fragment (16-bit 32x16): lane = column N, two contiguous 8-half chunks
// lanes 0-15: K = k0+0..7 and k0+16..23 ; lanes 16-31: K = k0+8..15 and k0+24..31
__device__ __forceinline__ v16h load_bfrag(const _Float16* colp, int lane) {
  const _Float16* p = colp + ((lane >> 4) << 3);  // +8 halves for upper half-wave
  v8h lo = *(const v8h*)(p);
  v8h hi = *(const v8h*)(p + 16);
  v16h b;
#pragma unroll
  for (int i = 0; i < 8; ++i) { b[i] = lo[i]; b[i + 8] = hi[i]; }
  return b;
}

__device__ __forceinline__ float sigf(float x) { return 1.0f / (1.0f + __expf(-x)); }

// ---------------- workspace layout (bytes) ----------------
#define XG_OFF   ((size_t)0)                      // f32 [NROWS][G4]   = 134217728
#define HALL_OFF ((size_t)134217728)              // f16 [NROWS][HID]  = 16777216
#define EM_OFF   ((size_t)150994944)              // f32 [NROWS][NTAGS]= 5242880
#define WIH_OFF  ((size_t)156237824)              // f16 [G4][KPAD]    = 229376
#define WHH_OFF  ((size_t)156467200)              // f16 [G4][HID]     = 131072
#define BSUM_OFF ((size_t)156598272)              // f32 [G4]          = 2048
#define NLL_OFF  ((size_t)156600320)              // f32 [BB]         = 512
#define WDEC_OFF ((size_t)156600832)              // f16 [32][HID]     = 8192

// ============ kernel 1: weight conversion / bias fold ============
__global__ void prep_kernel(const float* __restrict__ W_ih, const float* __restrict__ W_hh,
                            const float* __restrict__ b_ih, const float* __restrict__ b_hh,
                            const float* __restrict__ W_dec,
                            _Float16* __restrict__ wih_h, _Float16* __restrict__ whh_h,
                            float* __restrict__ bsum, _Float16* __restrict__ wdec_h) {
  int idx = blockIdx.x * blockDim.x + threadIdx.x;
  if (idx < G4 * KPAD) {
    int g = idx / KPAD, k = idx - g * KPAD;
    wih_h[idx] = (k < EMB) ? (_Float16)W_ih[g * EMB + k] : (_Float16)0.0f;
  }
  if (idx < G4 * HID) whh_h[idx] = (_Float16)W_hh[idx];
  if (idx < G4)       bsum[idx]  = b_ih[idx] + b_hh[idx];
  if (idx < 32 * HID) {
    int row = idx >> 7, k = idx & 127;
    wdec_h[idx] = (row < NTAGS) ? (_Float16)W_dec[row * HID + k] : (_Float16)0.0f;
  }
}

// ============ kernel 2: xg = embed(sentence) @ W_ih^T + (b_ih+b_hh) ============
// 1024 blocks x 256 threads; each block does 64 rows x 512 cols.
__global__ void xg_gemm_kernel(const int* __restrict__ sentence,
                               const float* __restrict__ embedding,
                               const _Float16* __restrict__ wih_h,
                               const float* __restrict__ bsum,
                               float* __restrict__ xg) {
  extern __shared__ char smem[];
  _Float16* a_tile = (_Float16*)smem;               // [64][KPAD]
  int*      s_idx  = (int*)(smem + 64 * KPAD * 2);  // [64]

  const int tid  = threadIdx.x;
  const int lane = tid & 31;
  const int wave = tid >> 5;
  const int row0 = blockIdx.x * 64;

  if (tid < 64) {
    int r = row0 + tid;                 // r = t*128 + b
    int t = r >> 7, b = r & 127;
    s_idx[tid] = sentence[b * SS + t];
  }
  __syncthreads();

  for (int i = tid; i < 64 * KPAD; i += 256) {
    int row = i / KPAD, k = i - row * KPAD;
    a_tile[i] = (k < EMB) ? (_Float16)embedding[(size_t)s_idx[row] * EMB + k]
                          : (_Float16)0.0f;
  }
  __syncthreads();

  // wave handles 4 n-tiles: nt = wave*4 .. wave*4+3 ; m-tiles 0..3
  for (int nti = 0; nti < 4; ++nti) {
    const int nt   = wave * 4 + nti;
    const int colN = nt * 16 + (lane & 15);
    const float bias = bsum[colN];
    for (int mt = 0; mt < 4; ++mt) {
      v8f acc = vzero8();
      const _Float16* arow = a_tile + (size_t)(mt * 16 + (lane & 15)) * KPAD;
#pragma unroll
      for (int kk = 0; kk < 7; ++kk) {
        v16h a = load_afrag(arow + kk * 32, lane);
        v16h b = load_bfrag(wih_h + (size_t)colN * KPAD + kk * 32, lane);
        acc = wmma16(a, b, acc);
      }
      const int rbase = row0 + mt * 16 + ((lane >> 4) << 3);
#pragma unroll
      for (int e = 0; e < 8; ++e)
        xg[(size_t)(rbase + e) * G4 + colN] = acc[e] + bias;
    }
  }
}

// ============ kernel 3: LSTM recurrence ============
// 8 blocks x 256 threads (8 waves). Block owns 16 batch rows; W_hh f16 resident in LDS,
// preloaded with gfx1250 async global->LDS DMA (ASYNCcnt).
__global__ void lstm_kernel(const float* __restrict__ xg,
                            const _Float16* __restrict__ whh_h,
                            _Float16* __restrict__ h_all) {
  extern __shared__ char smem[];
  _Float16* whh_s = (_Float16*)smem;             // [G4][HID] = 128 KB (offset 0)
  _Float16* hbuf  = (_Float16*)(smem + 131072);  // [16][HID] = 4 KB

  const int tid  = threadIdx.x;
  const int lane = tid & 31;
  const int nt   = tid >> 5;            // wave id = hidden 16-col tile (0..7)
  const int bb0  = blockIdx.x * 16;     // first batch row of this block

  // ---- async copy W_hh (f16, 128 KB) into LDS: 16 B per lane-op ----
  {
    const char* src = (const char*)whh_h;
    for (int i = tid * 16; i < G4 * HID * 2; i += 256 * 16) {
      unsigned loff = (unsigned)i;                 // whh_s sits at LDS offset 0
      const void* gp = (const void*)(src + i);
      asm volatile("global_load_async_to_lds_b128 %0, %1, off"
                   :: "v"(loff), "v"(gp) : "memory");
    }
    asm volatile("s_wait_asynccnt 0" ::: "memory");
  }
  for (int i = tid; i < 16 * HID; i += 256) hbuf[i] = (_Float16)0.0f;
  __syncthreads();

  const int colN = nt * 16 + (lane & 15);

  v8f cst = vzero8();

  for (int t = 0; t < SS; ++t) {
    // ---- G += h @ W_hh^T  (4 gates x 4 k-tiles of WMMA) ----
    v8f acc[4];
#pragma unroll
    for (int g = 0; g < 4; ++g) acc[g] = vzero8();

#pragma unroll
    for (int kk = 0; kk < 4; ++kk) {
      v16h a = load_afrag(hbuf + (size_t)(lane & 15) * HID + kk * 32, lane);
#pragma unroll
      for (int g = 0; g < 4; ++g) {
        v16h b = load_bfrag(whh_s + (size_t)(g * HID + colN) * HID + kk * 32, lane);
        acc[g] = wmma16(a, b, acc[g]);
      }
    }
    __syncthreads();   // all hbuf reads for step t done

    if (t + 1 < SS)
      __builtin_prefetch(xg + (size_t)((t + 1) * BB + bb0) * G4, 0, 1);

    // ---- gate nonlinearities, cell update, write h ----
    const size_t rbase = (size_t)(t * BB + bb0 + ((lane >> 4) << 3));
#pragma unroll
    for (int e = 0; e < 8; ++e) {
      const size_t ro = (rbase + e) * G4;
      float iv = acc[0][e] + xg[ro + 0 * HID + colN];
      float fv = acc[1][e] + xg[ro + 1 * HID + colN];
      float gv = acc[2][e] + xg[ro + 2 * HID + colN];
      float ov = acc[3][e] + xg[ro + 3 * HID + colN];
      float cn = sigf(fv) * cst[e] + sigf(iv) * tanhf(gv);
      float hv = sigf(ov) * tanhf(cn);
      cst[e] = cn;
      const int rowl = e + ((lane >> 4) << 3);
      hbuf[(size_t)rowl * HID + colN] = (_Float16)hv;
      h_all[(size_t)(t * BB + bb0 + rowl) * HID + colN] = (_Float16)hv;
    }
    __syncthreads();   // hbuf ready for step t+1
  }
}

// ============ kernel 4: emissions = h @ W_dec^T + b_dec  (WMMA, N padded 20->32) ========
// 512 blocks x 256 threads; wave = one 16-row m-tile, 2 n-tiles, K=128 (4 k-tiles).
__global__ void emis_kernel(const _Float16* __restrict__ h_all,
                            const _Float16* __restrict__ wdec_h,
                            const float* __restrict__ b_dec,
                            float* __restrict__ em) {
  const int tid  = threadIdx.x;
  const int lane = tid & 31;
  const int wave = tid >> 5;
  const int r0   = blockIdx.x * 128 + wave * 16;
  const _Float16* arow = h_all + (size_t)(r0 + (lane & 15)) * HID;

#pragma unroll
  for (int nt = 0; nt < 2; ++nt) {
    v8f acc = vzero8();
    const int colN = nt * 16 + (lane & 15);
#pragma unroll
    for (int kk = 0; kk < 4; ++kk) {
      v16h a = load_afrag(arow + kk * 32, lane);
      v16h b = load_bfrag(wdec_h + (size_t)colN * HID + kk * 32, lane);
      acc = wmma16(a, b, acc);
    }
    if (colN < NTAGS) {
      const float bias = b_dec[colN];
      const int rbase = r0 + ((lane >> 4) << 3);
#pragma unroll
      for (int e = 0; e < 8; ++e)
        em[(size_t)(rbase + e) * NTAGS + colN] = acc[e] + bias;
    }
  }
}

// ============ kernel 5: CRF NLL per batch (one wave per b) ============
__global__ void crf_kernel(const float* __restrict__ em,
                           const int* __restrict__ tags,
                           const unsigned char* __restrict__ mask,
                           const float* __restrict__ start_t,
                           const float* __restrict__ end_t,
                           const float* __restrict__ trans,
                           float* __restrict__ nll) {
  const int b    = blockIdx.x;
  const int lane = threadIdx.x;        // 32 threads, wave32
  const int jj   = (lane < NTAGS) ? lane : 0;
  const bool act = lane < NTAGS;

  float tcol[NTAGS];
#pragma unroll
  for (int i = 0; i < NTAGS; ++i) tcol[i] = trans[i * NTAGS + jj];

  float score = start_t[jj] + em[(size_t)b * NTAGS + jj];   // t=0 row index r=b

  float num = 0.0f;
  if (lane == 0) {
    int tag0 = tags[b * SS + 0];
    num = start_t[tag0] + em[(size_t)b * NTAGS + tag0];
  }

  for (int t = 1; t < SS; ++t) {
    const float m   = (float)mask[b * SS + t];
    const size_t rr = (size_t)(t * BB + b) * NTAGS;
    const float emj = act ? em[rr + jj] : 0.0f;

    float mmax = -INFINITY, ssum = 0.0f;
#pragma unroll
    for (int i = 0; i < NTAGS; ++i) {
      float si = __shfl(score, i, 32);
      float v  = si + tcol[i];
      float nm = fmaxf(mmax, v);
      ssum = ssum * __expf(mmax - nm) + __expf(v - nm);
      mmax = nm;
    }
    float nxt = mmax + __logf(ssum) + emj;
    score = (m > 0.5f) ? nxt : score;

    if (lane == 0 && m > 0.5f) {
      int tg = tags[b * SS + t], pt = tags[b * SS + t - 1];
      num += trans[pt * NTAGS + tg] + em[rr + tg];
    }
  }

  if (lane == 0) {
    int len = 0;
    for (int t = 0; t < SS; ++t) len += (int)mask[b * SS + t];
    num += end_t[tags[b * SS + len - 1]];
  }

  float fin = act ? (score + end_t[jj]) : -INFINITY;
  float mx = fin;
#pragma unroll
  for (int off = 16; off > 0; off >>= 1) mx = fmaxf(mx, __shfl_xor(mx, off, 32));
  float s = __expf(fin - mx);
#pragma unroll
  for (int off = 16; off > 0; off >>= 1) s += __shfl_xor(s, off, 32);
  float den = mx + __logf(s);

  if (lane == 0) nll[b] = num - den;
}

// ============ kernel 6: final mean reduce ============
__global__ void finalize_kernel(const float* __restrict__ nll, float* __restrict__ out) {
  __shared__ float red[BB];
  int t = threadIdx.x;
  red[t] = nll[t];
  __syncthreads();
  for (int s = 64; s > 0; s >>= 1) {
    if (t < s) red[t] += red[t + s];
    __syncthreads();
  }
  if (t == 0) out[0] = -red[0] / (float)BB;
}

// ================= host launcher =================
extern "C" void kernel_launch(void* const* d_in, const int* in_sizes, int n_in,
                              void* d_out, int out_size, void* d_ws, size_t ws_size,
                              hipStream_t stream) {
  const int*   sentence  = (const int*)  d_in[0];
  const int*   tags      = (const int*)  d_in[1];
  const unsigned char* mask = (const unsigned char*)d_in[2];
  const float* embedding = (const float*)d_in[3];
  const float* W_ih      = (const float*)d_in[4];
  const float* W_hh      = (const float*)d_in[5];
  const float* b_ih      = (const float*)d_in[6];
  const float* b_hh      = (const float*)d_in[7];
  const float* W_dec     = (const float*)d_in[8];
  const float* b_dec     = (const float*)d_in[9];
  const float* start_t   = (const float*)d_in[10];
  const float* end_t     = (const float*)d_in[11];
  const float* trans     = (const float*)d_in[12];
  float* out = (float*)d_out;

  char* ws = (char*)d_ws;
  float*    xg     = (float*)   (ws + XG_OFF);
  _Float16* h_all  = (_Float16*)(ws + HALL_OFF);
  float*    em     = (float*)   (ws + EM_OFF);
  _Float16* wih_h  = (_Float16*)(ws + WIH_OFF);
  _Float16* whh_h  = (_Float16*)(ws + WHH_OFF);
  float*    bsum   = (float*)   (ws + BSUM_OFF);
  float*    nll    = (float*)   (ws + NLL_OFF);
  _Float16* wdec_h = (_Float16*)(ws + WDEC_OFF);

  prep_kernel<<<(G4 * KPAD + 255) / 256, 256, 0, stream>>>(
      W_ih, W_hh, b_ih, b_hh, W_dec, wih_h, whh_h, bsum, wdec_h);

  xg_gemm_kernel<<<NROWS / 64, 256, 64 * KPAD * 2 + 64 * 4, stream>>>(
      sentence, embedding, wih_h, bsum, xg);

  lstm_kernel<<<BB / 16, 256, 131072 + 16 * HID * 2, stream>>>(xg, whh_h, h_all);

  emis_kernel<<<NROWS / 128, 256, 0, stream>>>(h_all, wdec_h, b_dec, em);

  crf_kernel<<<BB, 32, 0, stream>>>(em, tags, mask, start_t, end_t, trans, nll);

  finalize_kernel<<<1, BB, 0, stream>>>(nll, out);
}
